// LcRnnCell_81655918232337
// MI455X (gfx1250) — compile-verified, matrix operand order checked by reading, not run on hardware
//
#include <hip/hip_runtime.h>
#include <hip/hip_bf16.h>

// ---------------------------------------------------------------------------
// LC-RNN cell, B=4096, E=H=512, DEPTH=1, specialized for prev_depth in {0,1}:
//   gt0 = eqm = gtm = 0  =>  na01 = nb01 = 0, Wb01 GEMM is dead.
// 5 live GEMMs (26 GFLOP) run via v_wmma_f32_16x16x32_bf16.  Tile copies use
// GLOBAL_LOAD_ASYNC_TO_LDS_B128 (ASYNCcnt) when the toolchain exposes it,
// falling back to b128 load + ds_store.  All intermediates stay L2-resident
// (74 MB workspace < 192 MB L2), so no NT hints on stores.
// ---------------------------------------------------------------------------

#define B_ROWS 4096

typedef __attribute__((ext_vector_type(16))) __bf16 v16bf;
typedef __attribute__((ext_vector_type(8)))  __bf16 v8bf;
typedef __attribute__((ext_vector_type(8)))  float  v8f;
typedef int v4i_vs __attribute__((vector_size(16)));   // b128 payload type

#if defined(__has_builtin)
#  if __has_builtin(__builtin_amdgcn_global_load_async_to_lds_b128)
#    define USE_ASYNC_LDS 1
#  endif
#endif
#ifndef USE_ASYNC_LDS
#  define USE_ASYNC_LDS 0
#endif

#if USE_ASYNC_LDS
// clang prints the builtin's params as 'int4 __device__*' / '__shared__*':
// AS1 (global) source, AS3 (LDS) destination, imm offset, imm cpol.
#  define AS1_V4I(p) ((__attribute__((address_space(1))) v4i_vs*)(p))
#  define AS3_V4I(p) ((__attribute__((address_space(3))) v4i_vs*)(p))
#endif

__device__ __forceinline__ void wait_async_zero() {
#if USE_ASYNC_LDS
#  if __has_builtin(__builtin_amdgcn_s_wait_asynccnt)
  __builtin_amdgcn_s_wait_asynccnt(0);
#  else
  asm volatile("s_wait_asynccnt 0x0" ::: "memory");
#  endif
#endif
}

__device__ __forceinline__ __bf16 to_bf16(float f) {
  unsigned u = __builtin_bit_cast(unsigned, f);
  u += 0x7fffu + ((u >> 16) & 1u);                 // round-to-nearest-even
  unsigned short h = (unsigned short)(u >> 16);
  return __builtin_bit_cast(__bf16, h);
}

// Convert a (rows x cols) f32 slice (row stride/offset) into dense bf16.
__global__ __launch_bounds__(256) void convert_kernel(
    const float* __restrict__ src, __bf16* __restrict__ dst,
    int rows, int cols, int srcStride, int srcOff) {
  int idx = blockIdx.x * 256 + threadIdx.x;
  if (idx >= rows * cols) return;
  int r = idx / cols, c = idx - r * cols;
  dst[idx] = to_bf16(src[(size_t)r * srcStride + srcOff + c]);
}

__global__ __launch_bounds__(256) void masks_kernel(
    const int* __restrict__ pd, float* __restrict__ eq0,
    float* __restrict__ eqp, float* __restrict__ gtp, int n) {
  int i = blockIdx.x * 256 + threadIdx.x;
  if (i >= n) return;
  int d = pd[i];
  eq0[i] = (d == 1) ? 1.f : 0.f;   // pd == d      (d = 1)
  eqp[i] = (d == 0) ? 1.f : 0.f;   // pd + 1 == d
  gtp[i] = (d >= 1) ? 1.f : 0.f;   // pd + 1 >  d
}

// ---------------------------------------------------------------------------
// GEMM: out[m,n] = eq[m] * sum_k A[m,k] * W[n,k]  (+ gt[m] * addend[m,n])
// A is split in 512-wide column segments (concatenated activations).
// Tile 128x128x32, 8 wave32 waves (4M x 2N), wave tile 32x64 = 2x4 WMMA accs.
// ---------------------------------------------------------------------------
#define TM 128
#define TN 128
#define TK 32
#define LDSROW 40   // 32 bf16 + 8 pad (80B rows, keeps b128 alignment)

__global__ __launch_bounds__(256) void wmma_gemm_kernel(
    const __bf16* __restrict__ A0, const __bf16* __restrict__ A1,
    const __bf16* __restrict__ A2,
    const __bf16* __restrict__ W, int K,
    const float* __restrict__ eqMask, const float* __restrict__ gtMask,
    const float* __restrict__ addend, int addStride, int addOff,
    float* __restrict__ outF, __bf16* __restrict__ outBf) {

  __shared__ __attribute__((aligned(16))) __bf16 sA[2][TM * LDSROW];
  __shared__ __attribute__((aligned(16))) __bf16 sB[2][TN * LDSROW];

  const int mBase = blockIdx.y * TM;
  const int nBase = blockIdx.x * TN;
  const int tid   = threadIdx.x;
  const int wv    = tid >> 5;        // 8 wave32 waves
  const int lane  = tid & 31;
  const int waveM = wv & 3;          // 4 waves along M
  const int waveN = wv >> 2;         // 2 waves along N
  const int half  = lane >> 4;       // 16-bit WMMA K-half select
  const int lr    = lane & 15;

  auto loadTile = [&](int k0, int buf) {
    const int seg = k0 >> 9;                      // 512-wide segments
    const int kin = k0 & 511;
    const __bf16* Aseg = (seg == 0) ? A0 : ((seg == 1) ? A1 : A2);
    #pragma unroll
    for (int it = 0; it < 2; ++it) {
      int c   = tid + it * 256;                   // 512 chunks of 16B each
      int row = c >> 2;
      int col = (c & 3) * 8;
      const __bf16* ga = Aseg + (size_t)(mBase + row) * 512 + kin + col;
      const __bf16* gw = W + (size_t)(nBase + row) * K + k0 + col;
      __bf16* la = &sA[buf][row * LDSROW + col];
      __bf16* lb = &sB[buf][row * LDSROW + col];
#if USE_ASYNC_LDS
      // CDNA5 async DMA: global -> LDS, tracked with ASYNCcnt (no VGPR hop).
      __builtin_amdgcn_global_load_async_to_lds_b128(AS1_V4I(ga), AS3_V4I(la), 0, 0);
      __builtin_amdgcn_global_load_async_to_lds_b128(AS1_V4I(gw), AS3_V4I(lb), 0, 0);
#else
      *(v8bf*)la = *(const v8bf*)ga;
      *(v8bf*)lb = *(const v8bf*)gw;
      if (k0 + 2 * TK < K) {       // keep HBM pipe busy: global_prefetch_b8
        __builtin_prefetch(gw + 2 * TK, 0, 1);
      }
#endif
    }
  };

  loadTile(0, 0);

  v8f acc[2][4] = {};

  const int ksteps = K / TK;
  for (int ks = 0; ks < ksteps; ++ks) {
    wait_async_zero();               // this wave's async tile writes visible
    __syncthreads();                 // all waves' writes + prior reads done
    const int buf = ks & 1;
    if (ks + 1 < ksteps) loadTile((ks + 1) * TK, buf ^ 1);

    v16bf afrag[2], bfrag[4];
    // A 16x32 bf16 layout: lanes 0-15 hold K 0-7 & 16-23, lanes 16-31 hold
    // K 8-15 & 24-31 of row (lr). B uses the same per-lane pattern on W rows.
    #pragma unroll
    for (int i = 0; i < 2; ++i) {
      const __bf16* p = &sA[buf][(waveM * 32 + i * 16 + lr) * LDSROW + half * 8];
      v8bf lo = *(const v8bf*)p;
      v8bf hi = *(const v8bf*)(p + 16);
      afrag[i] = __builtin_shufflevector(lo, hi, 0,1,2,3,4,5,6,7,8,9,10,11,12,13,14,15);
    }
    #pragma unroll
    for (int j = 0; j < 4; ++j) {
      const __bf16* p = &sB[buf][(waveN * 64 + j * 16 + lr) * LDSROW + half * 8];
      v8bf lo = *(const v8bf*)p;
      v8bf hi = *(const v8bf*)(p + 16);
      bfrag[j] = __builtin_shufflevector(lo, hi, 0,1,2,3,4,5,6,7,8,9,10,11,12,13,14,15);
    }
    #pragma unroll
    for (int i = 0; i < 2; ++i)
      #pragma unroll
      for (int j = 0; j < 4; ++j)
        acc[i][j] = __builtin_amdgcn_wmma_f32_16x16x32_bf16(
            false, afrag[i], false, bfrag[j], (short)0, acc[i][j], false, false);
  }

  // Epilogue: C/D 16x16 f32 layout -> element (vgpr r, lane l) =
  // C[r + (l>=16)*8, l%16].  out = eq*acc (+ gt*addend), optional bf16 copy.
  #pragma unroll
  for (int i = 0; i < 2; ++i) {
    #pragma unroll
    for (int j = 0; j < 4; ++j) {
      #pragma unroll
      for (int r = 0; r < 8; ++r) {
        int row = mBase + waveM * 32 + i * 16 + half * 8 + r;
        int col = nBase + waveN * 64 + j * 16 + lr;
        float v = acc[i][j][r] * eqMask[row];
        if (gtMask)
          v += gtMask[row] * addend[(size_t)row * addStride + addOff + col];
        size_t o = (size_t)row * 512 + col;
        outF[o] = v;
        if (outBf) outBf[o] = to_bf16(v);
      }
    }
  }
}

// ---------------------------------------------------------------------------
// Phase 3: per-row attention logits (only ab10/ab11 slots are live; depth-0
// slots of every flat are zeros), softmax, mask, argmax-select, and outputs.
// One row per wave32; __shfl_xor width-32 reductions.
// ---------------------------------------------------------------------------
__global__ __launch_bounds__(256) void phase3_kernel(
    const float* __restrict__ na00, const float* __restrict__ na10,
    const float* __restrict__ nb00, const float* __restrict__ nb10,
    const float* __restrict__ nb11,
    const float* __restrict__ prev_a, const int* __restrict__ prev_depth,
    const float* __restrict__ Watt, float* __restrict__ out) {
  const int wv   = threadIdx.x >> 5;
  const int lane = threadIdx.x & 31;
  const int m    = blockIdx.x * 8 + wv;
  if (m >= B_ROWS) return;

  const int   pd  = prev_depth[m];
  const float eq0 = (pd == 1) ? 1.f : 0.f;
  const size_t r512  = (size_t)m * 512;
  const size_t r1024 = (size_t)m * 1024;

  float l[4] = {0.f, 0.f, 0.f, 0.f};
  for (int n = lane; n < 512; n += 32) {
    float va10 = na10[r512 + n];
    float vb10 = nb10[r512 + n];
    float va11 = eq0 * prev_a[r1024 + 512 + n];   // na11 = eq0 * a_d
    float vb11 = nb11[r512 + n];
    #pragma unroll
    for (int c = 0; c < 4; ++c) {
      const float* wr = Watt + c * 4096;
      l[c] += wr[1024 + n] * va10 + wr[1536 + n] * vb10 +
              wr[3072 + n] * va11 + wr[3584 + n] * vb11;
    }
  }
  #pragma unroll
  for (int c = 0; c < 4; ++c)
    for (int o = 16; o > 0; o >>= 1) l[c] += __shfl_xor(l[c], o, 32);

  // softmax over 4, mask, sel = sign(v - max) + 1 (ties -> 1 at every max)
  float mx = fmaxf(fmaxf(l[0], l[1]), fmaxf(l[2], l[3]));
  float e[4], s = 0.f;
  #pragma unroll
  for (int c = 0; c < 4; ++c) { e[c] = __expf(l[c] - mx); s += e[c]; }
  float inv = 1.f / s;
  float m0 = (pd == 0) ? 0.f : 1.f;   // 1 - e0
  float m2 = (pd == 1) ? 0.f : 1.f;   // 1 - ed (DEPTH == 1)
  float av[4] = { m0 * e[0] * inv, m0 * e[1] * inv,
                  m2 * e[2] * inv, m0 * e[3] * inv };
  float vm = fmaxf(fmaxf(av[0], av[1]), fmaxf(av[2], av[3]));
  float sel[4];
  #pragma unroll
  for (int c = 0; c < 4; ++c) sel[c] = (av[c] >= vm) ? 1.f : 0.f;

  float* next_a     = out;
  float* next_b     = out + (size_t)B_ROWS * 1024;
  float* next_depth = out + (size_t)2 * B_ROWS * 1024;
  float* fout       = next_depth + B_ROWS;
  float* jout       = fout + B_ROWS;

  for (int n = lane; n < 512; n += 32) {
    float na = sel[0] * na00[r512 + n] + sel[2] * na10[r512 + n] +
               sel[3] * (eq0 * prev_a[r1024 + 512 + n]);
    float nb = sel[0] * nb00[r512 + n] + sel[2] * nb10[r512 + n] +
               sel[3] * nb11[r512 + n];
    next_a[r1024 + n]       = 0.f;     // depth-0 slot of every branch is zero
    next_a[r1024 + 512 + n] = na;
    next_b[r1024 + n]       = 0.f;
    next_b[r1024 + 512 + n] = nb;
  }
  if (lane == 0) {
    float fpd = (float)pd;
    next_depth[m] = sel[0] * fpd + sel[1] * (fpd - 1.f) +
                    sel[2] * (fpd + 1.f) + sel[3] * fpd;
    fout[m] = sel[2] + sel[3];
    jout[m] = sel[1] + sel[3];
  }
}

// ---------------------------------------------------------------------------
extern "C" void kernel_launch(void* const* d_in, const int* in_sizes, int n_in,
                              void* d_out, int out_size, void* d_ws, size_t ws_size,
                              hipStream_t stream) {
  (void)in_sizes; (void)n_in; (void)out_size; (void)ws_size;

  const float* X          = (const float*)d_in[0];
  const float* prev_a     = (const float*)d_in[1];
  const float* prev_b     = (const float*)d_in[2];
  const int*   prev_depth = (const int*)d_in[3];
  const float* Wa00 = (const float*)d_in[4];
  const float* Wa10 = (const float*)d_in[5];
  const float* Wb00 = (const float*)d_in[6];
  const float* Wb11 = (const float*)d_in[7];
  const float* Wb10 = (const float*)d_in[8];
  // d_in[9] = Wb01: dead (eqm = gtm = 0 for DEPTH == 1)
  const float* Watt = (const float*)d_in[10];

  char* ws = (char*)d_ws;
  size_t off = 0;
  auto alloc = [&](size_t bytes) -> void* {
    void* p = ws + off;
    off += (bytes + 255) & ~(size_t)255;
    return p;
  };
  const size_t ACT = (size_t)B_ROWS * 512;
  __bf16* Xbf    = (__bf16*)alloc(ACT * 2);
  __bf16* a1bf   = (__bf16*)alloc(ACT * 2);
  __bf16* b0bf   = (__bf16*)alloc(ACT * 2);
  __bf16* b1bf   = (__bf16*)alloc(ACT * 2);
  __bf16* Wa00bf = (__bf16*)alloc((size_t)512 * 1536 * 2);
  __bf16* Wa10bf = (__bf16*)alloc((size_t)512 * 1024 * 2);
  __bf16* Wb00bf = (__bf16*)alloc((size_t)512 * 1536 * 2);
  __bf16* Wb11bf = (__bf16*)alloc((size_t)512 * 1024 * 2);
  __bf16* Wb10bf = (__bf16*)alloc((size_t)512 * 1024 * 2);
  __bf16* na00bf = (__bf16*)alloc(ACT * 2);
  __bf16* na10bf = (__bf16*)alloc(ACT * 2);
  float*  na00f  = (float*)alloc(ACT * 4);
  float*  na10f  = (float*)alloc(ACT * 4);
  float*  nb00f  = (float*)alloc(ACT * 4);
  float*  nb10f  = (float*)alloc(ACT * 4);
  float*  nb11f  = (float*)alloc(ACT * 4);
  float*  eq0    = (float*)alloc(B_ROWS * 4);
  float*  eqp    = (float*)alloc(B_ROWS * 4);
  float*  gtp    = (float*)alloc(B_ROWS * 4);

  const int actBlocks = (int)((ACT + 255) / 256);
  convert_kernel<<<actBlocks, 256, 0, stream>>>(X,      Xbf,  B_ROWS, 512,  512, 0);
  convert_kernel<<<actBlocks, 256, 0, stream>>>(prev_a, a1bf, B_ROWS, 512, 1024, 512);
  convert_kernel<<<actBlocks, 256, 0, stream>>>(prev_b, b0bf, B_ROWS, 512, 1024, 0);
  convert_kernel<<<actBlocks, 256, 0, stream>>>(prev_b, b1bf, B_ROWS, 512, 1024, 512);
  convert_kernel<<<(512 * 1536) / 256, 256, 0, stream>>>(Wa00, Wa00bf, 512, 1536, 1536, 0);
  convert_kernel<<<(512 * 1024) / 256, 256, 0, stream>>>(Wa10, Wa10bf, 512, 1024, 1024, 0);
  convert_kernel<<<(512 * 1536) / 256, 256, 0, stream>>>(Wb00, Wb00bf, 512, 1536, 1536, 0);
  convert_kernel<<<(512 * 1024) / 256, 256, 0, stream>>>(Wb11, Wb11bf, 512, 1024, 1024, 0);
  convert_kernel<<<(512 * 1024) / 256, 256, 0, stream>>>(Wb10, Wb10bf, 512, 1024, 1024, 0);
  masks_kernel<<<B_ROWS / 256, 256, 0, stream>>>(prev_depth, eq0, eqp, gtp, B_ROWS);

  dim3 ggrid(512 / TN, B_ROWS / TM);   // (4, 32)
  // Phase 1 (A side)
  wmma_gemm_kernel<<<ggrid, 256, 0, stream>>>(
      Xbf, b0bf, a1bf, Wa00bf, 1536, eq0, nullptr, nullptr, 0, 0, na00f, na00bf);
  wmma_gemm_kernel<<<ggrid, 256, 0, stream>>>(
      Xbf, b0bf, nullptr, Wa10bf, 1024, eqp, gtp, prev_a, 1024, 512, na10f, na10bf);
  // Phase 2 (B side, consumes na00/na10)
  wmma_gemm_kernel<<<ggrid, 256, 0, stream>>>(
      Xbf, a1bf, na00bf, Wb00bf, 1536, eq0, nullptr, nullptr, 0, 0, nb00f, nullptr);
  wmma_gemm_kernel<<<ggrid, 256, 0, stream>>>(
      Xbf, b1bf, nullptr, Wb11bf, 1024, eq0, nullptr, nullptr, 0, 0, nb11f, nullptr);
  wmma_gemm_kernel<<<ggrid, 256, 0, stream>>>(
      Xbf, na10bf, nullptr, Wb10bf, 1024, eqp, gtp, prev_b, 1024, 512, nb10f, nullptr);
  // Phase 3: attention + hard selection + outputs
  phase3_kernel<<<B_ROWS / 8, 256, 0, stream>>>(
      na00f, na10f, nb00f, nb10f, nb11f, prev_a, prev_depth, Watt, (float*)d_out);
}
